// QDRGAT_77309411473
// MI455X (gfx1250) — matrix operand order repaired and checked by reading, not compiled, not run on hardware
//
#include <hip/hip_runtime.h>

// ---------------------------------------------------------------------------
// QDRGAT forward for gfx1250 (MI455X), wave32 + v_wmma_f32_16x16x32_f16.
// BS=8, N=512, F=512, R=16, OUT=512, ALPHA=0.2
// ---------------------------------------------------------------------------

#define BS   8
#define NN   512
#define FF   512
#define RR   16
#define OUT  512

typedef _Float16 v16h __attribute__((ext_vector_type(16)));
typedef _Float16 v8h  __attribute__((ext_vector_type(8)));
typedef float    v8f  __attribute__((ext_vector_type(8)));

// ---------------------------------------------------------------------------
// Small batched GEMV: y[b,o] = act( sum_f x[b,f] * W[f,o] ), F_in=F_out=512.
// act: 0 = identity, 1 = ELU (alpha=1, jax.nn.elu default).
// grid (2, BS), block 256.
// ---------------------------------------------------------------------------
__global__ __launch_bounds__(256) void gemv_act(const float* __restrict__ x,
                                                const float* __restrict__ W,
                                                float* __restrict__ y, int act) {
  __shared__ float xs[FF];
  const int b = blockIdx.y;
  const int o = blockIdx.x * 256 + threadIdx.x;
  for (int f = threadIdx.x; f < FF; f += 256) xs[f] = x[(size_t)b * FF + f];
  __syncthreads();
  float acc = 0.f;
  for (int f = 0; f < FF; ++f) acc = fmaf(xs[f], W[(size_t)f * FF + o], acc);
  if (act == 1) acc = acc > 0.f ? acc : (__expf(acc) - 1.f);
  y[(size_t)b * FF + o] = acc;
}

// ---------------------------------------------------------------------------
// Pack x / x_ori to f16, plus transposed copy of x (B operand of attn@x).
// ---------------------------------------------------------------------------
__global__ __launch_bounds__(256) void pack_inputs(const float* __restrict__ x,
                                                   const float* __restrict__ x_ori,
                                                   _Float16* __restrict__ x_h,
                                                   _Float16* __restrict__ xori_h,
                                                   _Float16* __restrict__ xT_h) {
  size_t idx = (size_t)blockIdx.x * 256 + threadIdx.x;   // over BS*N*F
  int f = (int)(idx & 511);
  int n = (int)((idx >> 9) & 511);
  int b = (int)(idx >> 18);
  float vx = x[idx];
  x_h[idx]    = (_Float16)vx;
  xori_h[idx] = (_Float16)x_ori[idx];
  xT_h[((size_t)b << 18) + ((size_t)f << 9) + (size_t)n] = (_Float16)vx;
}

__global__ __launch_bounds__(256) void cvt_f32_f16(const float* __restrict__ in,
                                                   _Float16* __restrict__ out) {
  size_t idx = (size_t)blockIdx.x * 256 + threadIdx.x;
  out[idx] = (_Float16)in[idx];
}

// Wu [2F, OUT] row-major -> WuT [OUT, 2F] f16
__global__ __launch_bounds__(256) void transpose_wu(const float* __restrict__ Wu,
                                                    _Float16* __restrict__ WuT) {
  size_t idx = (size_t)blockIdx.x * 256 + threadIdx.x;   // over 1024*512
  int k2 = (int)(idx >> 9);
  int o  = (int)(idx & 511);
  WuT[(size_t)o * (2 * FF) + k2] = (_Float16)Wu[idx];
}

// ---------------------------------------------------------------------------
// WeffT[b][qk][r][k2] = sum_f Wv[k2,f] * ms[b,f] * Wa[qk*F + f, r]   (f16 out)
// grid (64, 2, BS), block 256: 16 k2 x 16 r per block.
// ---------------------------------------------------------------------------
__global__ __launch_bounds__(256) void weff_kernel(const float* __restrict__ Wqv,
                                                   const float* __restrict__ Wkv,
                                                   const float* __restrict__ mq,
                                                   const float* __restrict__ mk,
                                                   const float* __restrict__ Wa,
                                                   _Float16* __restrict__ WeffT) {
  __shared__ float was[FF * RR];   // 32 KB: ms[f]*Wa[f,r]
  const int b  = blockIdx.z;
  const int qk = blockIdx.y;
  const float* Wv = qk ? Wkv : Wqv;
  const float* ms = (qk ? mk : mq) + (size_t)b * FF;
  for (int idx = threadIdx.x; idx < FF * RR; idx += 256) {
    int f = idx >> 4, r = idx & 15;
    was[idx] = ms[f] * Wa[(size_t)(qk * FF + f) * RR + r];
  }
  __syncthreads();
  const int k2 = blockIdx.x * 16 + (threadIdx.x >> 4);
  const int r  = threadIdx.x & 15;
  const float* wrow = Wv + (size_t)k2 * FF;
  float acc = 0.f;
  for (int f = 0; f < FF; ++f) acc = fmaf(wrow[f], was[f * RR + r], acc);
  WeffT[((size_t)(b * 2 + qk) * RR + r) * (2 * FF) + k2] = (_Float16)acc;
}

// ---------------------------------------------------------------------------
// Fused scores + masked softmax, emits attn row as f16 (row-major GEMM A).
// grid (N, BS), block 256. One block per (b, i) row. Streams adj: 64B per j.
// ---------------------------------------------------------------------------
__global__ __launch_bounds__(256) void scores_softmax(const float* __restrict__ sq,
                                                      const float* __restrict__ sk,
                                                      const int* __restrict__ adj,
                                                      _Float16* __restrict__ attn) {
  __shared__ float sv[NN];
  __shared__ float red[256];
  const int b = blockIdx.y, i = blockIdx.x, tid = threadIdx.x;

  float sqi[RR];
  const float* sqp = sq + ((size_t)b * NN + i) * RR;
#pragma unroll
  for (int r = 0; r < RR; ++r) sqi[r] = sqp[r];

  // Prefetch the second half of this row's adjacency stream (gfx1250
  // global_prefetch_b8); adj is the dominant HBM consumer (134 MB).
  __builtin_prefetch(adj + (((size_t)b * NN + i) * NN + tid + 256) * RR, 0, 1);

  float lmax = -3.0e38f;
#pragma unroll
  for (int t = 0; t < 2; ++t) {
    const int j = tid + t * 256;
    const int* aj = adj + (((size_t)b * NN + i) * NN + j) * RR;
    const float* skp = sk + ((size_t)b * NN + j) * RR;
    float acc = 0.f, asum = 0.f;
#pragma unroll
    for (int r = 0; r < RR; ++r) {
      float a = (float)aj[r];
      float v = sqi[r] + skp[r];
      v = v > 0.f ? v : 0.2f * v;           // leaky_relu(alpha=0.2)
      acc  = fmaf(a, v, acc);
      asum += a;
    }
    float s = (asum > 0.f) ? acc : -9.0e15f;
    sv[j] = s;
    lmax = fmaxf(lmax, s);
  }
  red[tid] = lmax;
  __syncthreads();
  for (int off = 128; off > 0; off >>= 1) {
    if (tid < off) red[tid] = fmaxf(red[tid], red[tid + off]);
    __syncthreads();
  }
  const float mx = red[0];
  __syncthreads();
  float e0 = __expf(sv[tid] - mx);
  float e1 = __expf(sv[tid + 256] - mx);
  sv[tid] = e0;
  sv[tid + 256] = e1;
  red[tid] = e0 + e1;
  __syncthreads();
  for (int off = 128; off > 0; off >>= 1) {
    if (tid < off) red[tid] += red[tid + off];
    __syncthreads();
  }
  const float inv = 1.0f / red[0];
  _Float16* arow = attn + ((size_t)b * NN + i) * NN;
  arow[tid]       = (_Float16)(sv[tid] * inv);
  arow[tid + 256] = (_Float16)(sv[tid + 256] * inv);
}

// ---------------------------------------------------------------------------
// Batched f16 WMMA GEMM: C[f32, MxNc] = A[f16, MxK row-major] * B[f16, KxNc]
// with B supplied pre-transposed BT [Nc, K] row-major, and A optionally split
// at k=ksplit between two row-major sources (virtual concat along K).
//
// Fragment layouts per CDNA5 ISA 7.12.2:
//   A 16x32 f16 : lane row = lane&15; e<8 -> K = (lane>>4)*8 + e
//                                     e>=8 -> K = 16 + (lane>>4)*8 + (e-8)
//   B 32x16 f16 : lane col = lane&15; same K mapping as A (two 8-wide loads from BT)
//   C/D 16x16 f32: lane col = lane&15; rows = (lane>>4)*8 + i
//
// Template <MT, NT>: wave tile = (MT*16) x (NT*16).
//   NT==2: 4 waves arranged 2x2 -> block tile 128 x 64  (8 wmma / 6 loads)
//   NT==1: 4 waves stacked in M -> block tile 256 x 16  (narrow-N GEMMs)
//
// K loop: segmented per A source (no pointer select in the hot loop) and
// manually unrolled by 2 with ping-pong fragment register sets (E/O) so the
// compiler emits NO rotation copies and partial s_wait_loadcnt waits; the
// prefetch on the final pair is clamped to k0 (dead dummy load).
// Requires: every K segment a multiple of 64 (even # of 32-steps); all
// launches here use 512-wide segments (16 steps).
// ---------------------------------------------------------------------------
__device__ inline v16h load_a_frag(const _Float16* __restrict__ A, int lda,
                                   int mbase, int kbase, int lane) {
  const int row = lane & 15;
  const int kh  = lane >> 4;
  const _Float16* p = A + (size_t)(mbase + row) * lda + kbase + kh * 8;
  v8h lo = *(const v8h*)p;          // K = kh*8 + 0..7
  v8h hi = *(const v8h*)(p + 16);   // K = 16 + kh*8 + 0..7
  v16h a;
#pragma unroll
  for (int e = 0; e < 8; ++e) { a[e] = lo[e]; a[e + 8] = hi[e]; }
  return a;
}

__device__ inline v16h load_b_frag(const _Float16* __restrict__ BT, int ldb,
                                   int nbase, int kbase, int lane) {
  const int n  = lane & 15;
  const int kq = lane >> 4;
  const _Float16* p = BT + (size_t)(nbase + n) * ldb + kbase + kq * 8;
  v8h lo = *(const v8h*)p;
  v8h hi = *(const v8h*)(p + 16);
  v16h b;
#pragma unroll
  for (int e = 0; e < 8; ++e) { b[e] = lo[e]; b[e + 8] = hi[e]; }
  return b;
}

template <int MT, int NT>
__global__ __launch_bounds__(128) void wmma_gemm(
    const _Float16* __restrict__ A0, const _Float16* __restrict__ A1, int ksplit,
    long strideA, int lda,
    const _Float16* __restrict__ BT, long strideB, int ldb,
    float* __restrict__ C, long strideC, int ldc, int K) {
  constexpr int WN = (NT == 1) ? 1 : 2;   // waves along N
  constexpr int WM = 4 / WN;              // waves along M
  const int b = blockIdx.z;
  const _Float16* A0b = A0 + (size_t)b * strideA;
  const _Float16* A1b = A1 ? A1 + (size_t)b * strideA : A0b;
  const _Float16* BTb = BT + (size_t)b * strideB;
  float* Cb = C + (size_t)b * strideC;

  const int lane = threadIdx.x & 31;
  const int wave = threadIdx.x >> 5;
  const int wm = wave / WN;
  const int wn = wave % WN;
  const int mbase = blockIdx.y * (WM * MT * 16) + wm * MT * 16;
  const int nbase = blockIdx.x * (WN * NT * 16) + wn * NT * 16;

  v8f acc[MT][NT] = {};

  const _Float16* Aseg[2] = {A0b, A1b};
  const int segk0[2] = {0, ksplit};
  const int segk1[2] = {(ksplit < K) ? ksplit : K, K};
#pragma unroll
  for (int s = 0; s < 2; ++s) {
    const _Float16* Ap = Aseg[s];
    const int k0 = segk0[s];
    const int k1 = segk1[s];
    if (k0 >= k1) continue;                // uniform across the wave
    const int nsteps = (k1 - k0) >> 5;     // even by construction

    v16h bE[NT], aE[MT], bO[NT], aO[MT];
#pragma unroll
    for (int j = 0; j < NT; ++j) bE[j] = load_b_frag(BTb, ldb, nbase + 16 * j, k0, lane);
#pragma unroll
    for (int i = 0; i < MT; ++i) aE[i] = load_a_frag(Ap, lda, mbase + 16 * i, 0, lane);

    for (int it = 0; it < nsteps; it += 2) {
      // Prefetch odd step, consume even set.
      const int kb1 = k0 + (it + 1) * 32;
#pragma unroll
      for (int j = 0; j < NT; ++j)
        bO[j] = load_b_frag(BTb, ldb, nbase + 16 * j, kb1, lane);
#pragma unroll
      for (int i = 0; i < MT; ++i)
        aO[i] = load_a_frag(Ap, lda, mbase + 16 * i, kb1 - k0, lane);
#pragma unroll
      for (int i = 0; i < MT; ++i)
#pragma unroll
        for (int j = 0; j < NT; ++j)
          acc[i][j] = __builtin_amdgcn_wmma_f32_16x16x32_f16(
              false, aE[i], false, bE[j], (short)0, acc[i][j], false, false);
      // Prefetch next even step (clamped dummy on final pair), consume odd.
      const int kb2 = (it + 2 < nsteps) ? k0 + (it + 2) * 32 : k0;
#pragma unroll
      for (int j = 0; j < NT; ++j)
        bE[j] = load_b_frag(BTb, ldb, nbase + 16 * j, kb2, lane);
#pragma unroll
      for (int i = 0; i < MT; ++i)
        aE[i] = load_a_frag(Ap, lda, mbase + 16 * i, kb2 - k0, lane);
#pragma unroll
      for (int i = 0; i < MT; ++i)
#pragma unroll
        for (int j = 0; j < NT; ++j)
          acc[i][j] = __builtin_amdgcn_wmma_f32_16x16x32_f16(
              false, aO[i], false, bO[j], (short)0, acc[i][j], false, false);
    }
  }

  const int n0   = lane & 15;
  const int mrow = (lane >> 4) * 8;
#pragma unroll
  for (int i = 0; i < MT; ++i)
#pragma unroll
    for (int j = 0; j < NT; ++j) {
      float* cp = Cb + (size_t)(mbase + i * 16 + mrow) * ldc + nbase + j * 16 + n0;
#pragma unroll
      for (int e = 0; e < 8; ++e) cp[(size_t)e * ldc] = acc[i][j][e];
    }
}

// ---------------------------------------------------------------------------
// Host-side orchestration.
// ---------------------------------------------------------------------------
extern "C" void kernel_launch(void* const* d_in, const int* in_sizes, int n_in,
                              void* d_out, int out_size, void* d_ws, size_t ws_size,
                              hipStream_t stream) {
  const float* x_ori = (const float*)d_in[0];
  const float* x     = (const float*)d_in[1];
  const float* c     = (const float*)d_in[2];
  const int*   adj   = (const int*)  d_in[3];
  const float* Wfc   = (const float*)d_in[4];
  const float* Wdc   = (const float*)d_in[5];
  const float* Wqv   = (const float*)d_in[6];
  const float* Wkv   = (const float*)d_in[7];
  const float* Wqc   = (const float*)d_in[8];
  const float* Wkc   = (const float*)d_in[9];
  const float* Wa    = (const float*)d_in[10];
  const float* Wu    = (const float*)d_in[11];
  float* out = (float*)d_out;

  // Workspace carve-up (256B aligned).
  char* ws = (char*)d_ws;
  size_t off = 0;
  auto carve = [&](size_t bytes) -> void* {
    void* p = ws + off;
    off = (off + bytes + 255) & ~(size_t)255;
    return p;
  };
  const size_t BNF = (size_t)BS * NN * FF;           // 2,097,152 elements
  float*    t_buf   = (float*)   carve((size_t)BS * FF * 4);
  float*    m_buf   = (float*)   carve((size_t)BS * FF * 4);
  float*    mq_buf  = (float*)   carve((size_t)BS * FF * 4);
  float*    mk_buf  = (float*)   carve((size_t)BS * FF * 4);
  _Float16* x_h     = (_Float16*)carve(BNF * 2);
  _Float16* xori_h  = (_Float16*)carve(BNF * 2);
  _Float16* xT_h    = (_Float16*)carve(BNF * 2);
  _Float16* newx_h  = (_Float16*)carve(BNF * 2);
  float*    newx_f  = (float*)   carve(BNF * 4);
  _Float16* attn_h  = (_Float16*)carve((size_t)BS * NN * NN * 2);
  _Float16* WeffT   = (_Float16*)carve((size_t)BS * 2 * RR * (2 * FF) * 2);
  float*    sq_buf  = (float*)   carve((size_t)BS * NN * RR * 4);
  float*    sk_buf  = (float*)   carve((size_t)BS * NN * RR * 4);
  _Float16* WuT     = (_Float16*)carve((size_t)OUT * (2 * FF) * 2);
  (void)ws_size; (void)in_sizes; (void)n_in; (void)out_size;

  // 1) m-chain: t = elu(c@Wfc); m = t@Wdc; mq = m@Wqc; mk = m@Wkc
  dim3 gvGrid(FF / 256, BS);
  gemv_act<<<gvGrid, 256, 0, stream>>>(c,     Wfc, t_buf,  1);
  gemv_act<<<gvGrid, 256, 0, stream>>>(t_buf, Wdc, m_buf,  0);
  gemv_act<<<gvGrid, 256, 0, stream>>>(m_buf, Wqc, mq_buf, 0);
  gemv_act<<<gvGrid, 256, 0, stream>>>(m_buf, Wkc, mk_buf, 0);

  // 2) f16 packs / transposes
  pack_inputs<<<(unsigned)(BNF / 256), 256, 0, stream>>>(x, x_ori, x_h, xori_h, xT_h);
  transpose_wu<<<(unsigned)((size_t)2 * FF * OUT / 256), 256, 0, stream>>>(Wu, WuT);

  // 3) per-batch effective attention weights (folds Wqv*diag(m@Wqc)*Wa[:F])
  weff_kernel<<<dim3(2 * FF / 16, 2, BS), 256, 0, stream>>>(Wqv, Wkv, mq_buf,
                                                            mk_buf, Wa, WeffT);

  // 4) sq = [x|x_ori] @ Weff_q[b],  sk = [x|x_ori] @ Weff_k[b]   (WMMA <4,1>)
  //    M=512, K=1024, Nc=16; A split at k=512 between x and x_ori.
  wmma_gemm<4, 1><<<dim3(1, NN / 256, BS), 128, 0, stream>>>(
      x_h, xori_h, FF, (long)NN * FF, FF,
      WeffT,               (long)2 * RR * (2 * FF), 2 * FF,
      sq_buf, (long)NN * RR, RR, 2 * FF);
  wmma_gemm<4, 1><<<dim3(1, NN / 256, BS), 128, 0, stream>>>(
      x_h, xori_h, FF, (long)NN * FF, FF,
      WeffT + (size_t)RR * (2 * FF), (long)2 * RR * (2 * FF), 2 * FF,
      sk_buf, (long)NN * RR, RR, 2 * FF);

  // 5) fused scores + masked softmax -> attn (f16, row-major)
  scores_softmax<<<dim3(NN, BS), 256, 0, stream>>>(sq_buf, sk_buf, adj, attn_h);

  // 6) new_x = attn @ x   (WMMA <4,2>; B = xT)  M=N=K=512
  wmma_gemm<4, 2><<<dim3(NN / 64, NN / 128, BS), 128, 0, stream>>>(
      attn_h, nullptr, NN, (long)NN * NN, NN,
      xT_h, (long)NN * FF, NN,
      newx_f, (long)NN * FF, FF, NN);
  cvt_f32_f16<<<(unsigned)(BNF / 256), 256, 0, stream>>>(newx_f, newx_h);

  // 7) out = [x | new_x] @ Wu   (WMMA <4,2>; A split at k=512; BT = WuT shared)
  wmma_gemm<4, 2><<<dim3(OUT / 64, NN / 128, BS), 128, 0, stream>>>(
      x_h, newx_h, FF, (long)NN * FF, FF,
      WuT, 0L, 2 * FF,
      out, (long)NN * OUT, OUT, 2 * FF);
}
